// SimpleGIN_9208409883073
// MI455X (gfx1250) — compile-verified, matrix operand order checked by reading, not run on hardware
//
#include <hip/hip_runtime.h>
#include <hip/hip_bf16.h>

#define N_NODES 50000
#define DIM     128
#define DEG     16
#define LN_EPS  1e-5f

// LDS pitches (in elements). 136 halves = 272B = 17*16B -> v8h loads stay
// 16B-aligned per row and lane->bank mapping (68 dwords/row) is conflict-free.
#define PA 136
#define PW 136
#define PY 132

typedef __attribute__((ext_vector_type(16))) _Float16 v16h;
typedef __attribute__((ext_vector_type(8)))  _Float16 v8h;
typedef __attribute__((ext_vector_type(4)))  _Float16 h4;
typedef __attribute__((ext_vector_type(8)))  float    v8f;

// ---------------------------------------------------------------------------
// Convert W[l] (fp32, [fi][fo] row-major) into Wt[l] (f16, [fo][fi]) so that
// B-fragment loads are K-contiguous, mirroring the A-fragment layout.
// ---------------------------------------------------------------------------
__global__ __launch_bounds__(256) void prep_weights(
    const float* __restrict__ W1, const float* __restrict__ W2,
    const float* __restrict__ W3, _Float16* __restrict__ wt) {
  int idx = blockIdx.x * 256 + threadIdx.x;      // 3*128*128 = 49152 total
  int l = idx >> 14;
  int r = idx & 16383;
  int n = r >> 7;          // output feature
  int k = r & 127;         // input feature
  const float* W = (l == 0) ? W1 : ((l == 1) ? W2 : W3);
  wt[idx] = (_Float16)W[k * DIM + n];            // wt[l][n][k] = W[k][n]
}

// ---------------------------------------------------------------------------
// GIN 'max' aggregation + (1+eps)*h + agg, emitted as f16 rows.
// One wave per node; lane L handles float4 columns [4L..4L+3].
// In-neighbors of node v are edge_src[16v..16v+15] plus the self loop (v).
// ---------------------------------------------------------------------------
__global__ __launch_bounds__(256) void agg_convert(
    const float* __restrict__ h, const int* __restrict__ esrc,
    const float* __restrict__ epsp, _Float16* __restrict__ zh) {
  int wave = threadIdx.x >> 5;
  int lane = threadIdx.x & 31;
  int v = blockIdx.x * 8 + wave;                 // 6250 blocks * 8 waves = 50000

  const float4* hv4 = (const float4*)(h + (size_t)v * DIM);
  float4 hv  = hv4[lane];
  float4 acc = hv;                               // self loop participates in max
  const int* e = esrc + v * DEG;
#pragma unroll
  for (int j = 0; j < DEG; j++) {
    int s = e[j];
    float4 t = ((const float4*)(h + (size_t)s * DIM))[lane];
    acc.x = fmaxf(acc.x, t.x);
    acc.y = fmaxf(acc.y, t.y);
    acc.z = fmaxf(acc.z, t.z);
    acc.w = fmaxf(acc.w, t.w);
  }
  float ep = 1.0f + *epsp;
  h4 z;
  z[0] = (_Float16)(ep * hv.x + acc.x);
  z[1] = (_Float16)(ep * hv.y + acc.y);
  z[2] = (_Float16)(ep * hv.z + acc.z);
  z[3] = (_Float16)(ep * hv.w + acc.w);
  ((h4*)(zh + (size_t)v * DIM))[lane] = z;
}

// ---------------------------------------------------------------------------
// Per 16-node block: y = z @ W + b via WMMA f16->f32, then LayerNorm + ELU.
// 128 threads = 4 waves; wave w computes output column tiles {w, w+4},
// each a 16x16 tile accumulated over K=128 in 4 WMMA steps.
// ---------------------------------------------------------------------------
__global__ __launch_bounds__(128) void gemm_ln_elu(
    const _Float16* __restrict__ zh, const _Float16* __restrict__ wtl,
    const float* __restrict__ bias, const float* __restrict__ gamma,
    const float* __restrict__ beta, float* __restrict__ out) {
  __shared__ __align__(16) _Float16 As[16 * PA];
  __shared__ __align__(16) _Float16 Ws[128 * PW];
  __shared__ float Ys[16 * PY];

  int tid   = threadIdx.x;
  int node0 = blockIdx.x * 16;

  // Stage A tile: 16 rows x 128 halves (1024 dwords), coalesced.
  const uint* asrc = (const uint*)(zh + (size_t)node0 * DIM);
#pragma unroll
  for (int i = 0; i < 8; i++) {
    int d = tid + i * 128;
    int r = d >> 6, c2 = d & 63;
    *(uint*)&As[r * PA + c2 * 2] = asrc[d];
  }
  // Stage W^T: 128 rows x 128 halves (8192 dwords), coalesced.
  const uint* wsrc = (const uint*)wtl;
#pragma unroll 4
  for (int i = 0; i < 64; i++) {
    int d = tid + i * 128;
    int r = d >> 6, c2 = d & 63;
    *(uint*)&Ws[r * PW + c2 * 2] = wsrc[d];
  }
  __syncthreads();

  int wave = tid >> 5, lane = tid & 31;
  int m  = lane & 15;        // A row (lo/hi halves share rows), also B/C column
  int hi = lane >> 4;        // selects K window 8..15 / 24..31 per ISA layout
  int n0a = wave, n0b = wave + 4;

  float bva = bias[n0a * 16 + m];
  float bvb = bias[n0b * 16 + m];
  v8f c0, c1;
#pragma unroll
  for (int r = 0; r < 8; r++) { c0[r] = bva; c1[r] = bvb; }

#pragma unroll
  for (int kb = 0; kb < 4; kb++) {
    int kbase = kb * 32 + hi * 8;
    v8h alo = *(const v8h*)&As[m * PA + kbase];
    v8h ahi = *(const v8h*)&As[m * PA + kbase + 16];
    v8h b0l = *(const v8h*)&Ws[(n0a * 16 + m) * PW + kbase];
    v8h b0h = *(const v8h*)&Ws[(n0a * 16 + m) * PW + kbase + 16];
    v8h b1l = *(const v8h*)&Ws[(n0b * 16 + m) * PW + kbase];
    v8h b1h = *(const v8h*)&Ws[(n0b * 16 + m) * PW + kbase + 16];
    v16h a, b0, b1;
#pragma unroll
    for (int i = 0; i < 8; i++) {
      a[i] = alo[i];  a[i + 8] = ahi[i];
      b0[i] = b0l[i]; b0[i + 8] = b0h[i];
      b1[i] = b1l[i]; b1[i + 8] = b1h[i];
    }
    c0 = __builtin_amdgcn_wmma_f32_16x16x32_f16(false, a, false, b0,
                                                (short)0, c0, false, false);
    c1 = __builtin_amdgcn_wmma_f32_16x16x32_f16(false, a, false, b1,
                                                (short)0, c1, false, false);
  }

  // C layout: element r -> row (r + 8*hi), col (lane&15) within the tile.
#pragma unroll
  for (int r = 0; r < 8; r++) {
    int row = r + hi * 8;
    Ys[row * PY + n0a * 16 + m] = c0[r];
    Ys[row * PY + n0b * 16 + m] = c1[r];
  }
  __syncthreads();

  // LayerNorm over 128 features + ELU. 8 threads per row, 16 cols each.
  int row = tid >> 3, sub = tid & 7;
  float vals[16];
  float s = 0.f, ss = 0.f;
#pragma unroll
  for (int i = 0; i < 16; i++) {
    float y = Ys[row * PY + sub * 16 + i];
    vals[i] = y;
    s += y; ss += y * y;
  }
#pragma unroll
  for (int o = 1; o < 8; o <<= 1) {
    s  += __shfl_xor(s,  o, 8);
    ss += __shfl_xor(ss, o, 8);
  }
  float mu  = s * (1.f / 128.f);
  float var = ss * (1.f / 128.f) - mu * mu;
  float inv = rsqrtf(var + LN_EPS);
  float* op = out + (size_t)(node0 + row) * DIM + sub * 16;
#pragma unroll
  for (int i = 0; i < 16; i++) {
    int col = sub * 16 + i;
    float y = (vals[i] - mu) * inv * gamma[col] + beta[col];
    op[i] = (y > 0.f) ? y : expm1f(y);
  }
}

// ---------------------------------------------------------------------------
extern "C" void kernel_launch(void* const* d_in, const int* in_sizes, int n_in,
                              void* d_out, int out_size, void* d_ws, size_t ws_size,
                              hipStream_t stream) {
  const float* x    = (const float*)d_in[0];
  const int*   esrc = (const int*)d_in[1];
  // d_in[2] = edge_dst: implied by construction (edge i<N*DEG -> dst=i/16,
  // self-loops appended), so it is not needed.
  const float* W1 = (const float*)d_in[3];
  const float* b1 = (const float*)d_in[4];
  const float* W2 = (const float*)d_in[5];
  const float* b2 = (const float*)d_in[6];
  const float* W3 = (const float*)d_in[7];
  const float* b3 = (const float*)d_in[8];
  const float* gamma = (const float*)d_in[9];
  const float* beta  = (const float*)d_in[10];
  const float* eps   = (const float*)d_in[11];

  // Workspace layout (~64.1 MB total):
  //   zh: f16 [N][128]           12,800,000 B
  //   wt: f16 [3][128][128]          98,304 B
  //   hA: f32 [N][128]           25,600,000 B
  //   hB: f32 [N][128]           25,600,000 B
  char* ws = (char*)d_ws;
  const size_t ZH = (size_t)N_NODES * DIM * 2;
  const size_t WT = (size_t)3 * DIM * DIM * 2;
  _Float16* zh = (_Float16*)ws;
  _Float16* wt = (_Float16*)(ws + ZH);
  float* hA = (float*)(ws + ZH + WT);
  float* hB = hA + (size_t)N_NODES * DIM;

  prep_weights<<<192, 256, 0, stream>>>(W1, W2, W3, wt);

  const float* hin[3]  = {x, hA, hB};
  float*       hout[3] = {hA, hB, (float*)d_out};
  const float* bl[3]   = {b1, b2, b3};

  for (int l = 0; l < 3; l++) {
    agg_convert<<<N_NODES / 8, 256, 0, stream>>>(hin[l], esrc, eps + l, zh);
    gemm_ln_elu<<<N_NODES / 16, 128, 0, stream>>>(
        zh, wt + (size_t)l * DIM * DIM, bl[l],
        gamma + l * DIM, beta + l * DIM, hout[l]);
  }
}
